// Attention_45372034515489
// MI455X (gfx1250) — matrix operand order repaired and checked by reading, not compiled, not run on hardware
//
#include <hip/hip_runtime.h>
#include <hip/hip_bf16.h>

typedef __bf16 bf16_t;
typedef __attribute__((ext_vector_type(16))) __bf16 bf16x16;
typedef __attribute__((ext_vector_type(8)))  __bf16 bf16x8;
typedef __attribute__((ext_vector_type(4)))  __bf16 bf16x4;
typedef __attribute__((ext_vector_type(8)))  float  f32x8;

#define B_SZ        4
#define T_SZ        2048
#define D_MODEL_C   2048
#define NUM_HEADS_C 16
#define QGROUPS     4
#define HEAD_DIM_C  128
#define QKV_N       3072
#define SCALE_C     0.08838834764831845f
#define LOG_THETA   9.210340371976184f   // ln(10000)

// ---------------------------------------------------------------------------
// WMMA helpers (CDNA5 16x16x32 bf16, f32 accumulate)
// ---------------------------------------------------------------------------
__device__ __forceinline__ f32x8 wmma_bf16(bf16x16 a, bf16x16 b, f32x8 c) {
  // 8 args: (neg_a, A, neg_b, B, c_mod, C, reuse_a, reuse_b)
  return __builtin_amdgcn_wmma_f32_16x16x32_bf16(false, a, false, b,
                                                 (short)0, c, false, false);
}

// A-matrix fragment (16x32 bf16, row-major source, row stride in elements).
// Lane m (0-15): row m, K = {0..7, 16..23}. Lane m+16: row m, K = {8..15, 24..31}.
__device__ __forceinline__ bf16x16 load_frag_a(const bf16_t* base, int stride) {
  const int lane = threadIdx.x & 31;
  const bf16_t* p = base + (lane & 15) * stride + ((lane >> 4) << 3);
  union { bf16x16 v; bf16x8 h[2]; } u;
  u.h[0] = *(const bf16x8*)(p);
  u.h[1] = *(const bf16x8*)(p + 16);
  return u.v;
}

// B-matrix fragment (32x16 bf16) loaded from an [N][K] (transposed) tile.
// Lane n (0-15): column n, K = {0..7, 16..23}. Lane n+16: column n, K = {8..15, 24..31}.
__device__ __forceinline__ bf16x16 load_frag_b(const bf16_t* base, int stride) {
  const int lane = threadIdx.x & 31;
  const bf16_t* p = base + (lane & 15) * stride + ((lane >> 4) << 3);
  union { bf16x16 v; bf16x8 h[2]; } u;
  u.h[0] = *(const bf16x8*)(p);
  u.h[1] = *(const bf16x8*)(p + 16);
  return u.v;
}

// ---------------------------------------------------------------------------
// fp32 -> bf16 convert (grid-stride)
// ---------------------------------------------------------------------------
__global__ void cvt_f32_bf16(const float* __restrict__ s, bf16_t* __restrict__ d,
                             size_t n) {
  size_t i = (size_t)blockIdx.x * blockDim.x + threadIdx.x;
  const size_t stride = (size_t)gridDim.x * blockDim.x;
  for (; i < n; i += stride) d[i] = (bf16_t)s[i];
}

// ---------------------------------------------------------------------------
// Tiled bf16 GEMM:  C[M x N] (f32) = A[M x K] * B[K x N]   (row-major)
// N, K are compile-time so the epilogue folds row strides into immediate
// store offsets. Block: 256 threads = 8 waves; tile 128(M) x 128(N) x 32(K).
// Wave grid 4x2; each wave computes a 32x64 sub-tile via 2x4 WMMA frags
// (8 v_wmma per K-step).
// ---------------------------------------------------------------------------
template <int N, int K>
__global__ __launch_bounds__(256)
void gemm_bf16_f32(const bf16_t* __restrict__ A, const bf16_t* __restrict__ Bm,
                   float* __restrict__ C) {
  alignas(32) __shared__ bf16_t As[128][40];  // 128x32 + pad
  alignas(32) __shared__ bf16_t Bs[128][40];  // [n][k] transposed tile + pad
  const int tid  = threadIdx.x;
  const int wave = tid >> 5;
  const int lane = tid & 31;
  const int bm = blockIdx.x * 128;
  const int bn = blockIdx.y * 128;
  const int wm = (wave >> 1) * 32;   // 0,32,64,96
  const int wn = (wave & 1) * 64;    // 0,64
  f32x8 acc[2][4] = {};
  const int ra   = tid >> 2;         // 0..63
  const int ca   = (tid & 3) * 8;    // 0,8,16,24
  const int kb_r = tid >> 5;         // 0..7
  const int nb4  = (tid & 31) * 4;   // 0..124

  for (int k0 = 0; k0 < K; k0 += 32) {
    // A tile: 128x32, 8 bf16 (16B) per thread per pass, 2 passes
    *(bf16x8*)&As[ra][ca]      = *(const bf16x8*)&A[(size_t)(bm + ra) * K + k0 + ca];
    *(bf16x8*)&As[ra + 64][ca] = *(const bf16x8*)&A[(size_t)(bm + ra + 64) * K + k0 + ca];
    // B tile transposed: Bs[n][k] = B[(k0+k)*N + bn+n], vector global reads
#pragma unroll
    for (int p = 0; p < 4; ++p) {
      const int k = kb_r + p * 8;
      bf16x4 t = *(const bf16x4*)&Bm[(size_t)(k0 + k) * N + bn + nb4];
#pragma unroll
      for (int j = 0; j < 4; ++j) Bs[nb4 + j][k] = t[j];
    }
    if (k0 + 32 < K) {  // prefetch next K tile (global_prefetch_b8)
      __builtin_prefetch(&A[(size_t)(bm + ra) * K + k0 + 32 + ca], 0, 1);
      __builtin_prefetch(&Bm[(size_t)(k0 + 32 + kb_r) * N + bn + nb4], 0, 1);
    }
    __syncthreads();

    bf16x16 af0 = load_frag_a(&As[wm][0], 40);
    bf16x16 af1 = load_frag_a(&As[wm + 16][0], 40);
#pragma unroll
    for (int j = 0; j < 4; ++j) {
      bf16x16 bfj = load_frag_b(&Bs[wn + j * 16][0], 40);
      acc[0][j] = wmma_bf16(af0, bfj, acc[0][j]);
      acc[1][j] = wmma_bf16(af1, bfj, acc[1][j]);
    }
    __syncthreads();
  }

  const int half8 = (lane >> 4) << 3;  // C-frag: upper lanes own rows +8
#pragma unroll
  for (int i = 0; i < 2; ++i)
#pragma unroll
    for (int j = 0; j < 4; ++j) {
      float* cp = C + (size_t)(bm + wm + i * 16 + half8) * N
                    + (bn + wn + j * 16 + (lane & 15));
#pragma unroll
      for (int r = 0; r < 8; ++r)
        cp[(size_t)r * N] = acc[i][j][r];   // constant offsets (N is constexpr)
    }
}

// ---------------------------------------------------------------------------
// RoPE (f32) + repack qkv -> head-major bf16 Q[B,H,T,D], K/V[B,G,T,D]
// One block per (b,t) row of qkv.
// ---------------------------------------------------------------------------
__global__ __launch_bounds__(256)
void rope_pack(const float* __restrict__ qkv, bf16_t* __restrict__ qb,
               bf16_t* __restrict__ kb, bf16_t* __restrict__ vb) {
  const int row = blockIdx.x;          // b*T + t
  const int b = row >> 11;
  const int t = row & 2047;
  const float* src = qkv + (size_t)row * QKV_N;
  const float tf = (float)t;

  for (int p = threadIdx.x; p < NUM_HEADS_C * 64; p += blockDim.x) {
    const int h = p >> 6, i = p & 63;
    const float inv = __expf(-(float)i * (LOG_THETA / 64.f));
    float s, c; __sincosf(tf * inv, &s, &c);
    const float x1 = src[h * HEAD_DIM_C + i];
    const float x2 = src[h * HEAD_DIM_C + 64 + i];
    const size_t o = ((size_t)(b * NUM_HEADS_C + h) * T_SZ + t) * HEAD_DIM_C;
    qb[o + i]      = (bf16_t)(x1 * c - x2 * s);
    qb[o + 64 + i] = (bf16_t)(x1 * s + x2 * c);
  }
  for (int p = threadIdx.x; p < QGROUPS * 64; p += blockDim.x) {
    const int g = p >> 6, i = p & 63;
    const float inv = __expf(-(float)i * (LOG_THETA / 64.f));
    float s, c; __sincosf(tf * inv, &s, &c);
    const float x1 = src[D_MODEL_C + g * HEAD_DIM_C + i];
    const float x2 = src[D_MODEL_C + g * HEAD_DIM_C + 64 + i];
    const size_t o = ((size_t)(b * QGROUPS + g) * T_SZ + t) * HEAD_DIM_C;
    kb[o + i]      = (bf16_t)(x1 * c - x2 * s);
    kb[o + 64 + i] = (bf16_t)(x1 * s + x2 * c);
  }
  for (int e = threadIdx.x; e < QGROUPS * HEAD_DIM_C; e += blockDim.x) {
    const int g = e >> 7, d = e & 127;
    const size_t o = ((size_t)(b * QGROUPS + g) * T_SZ + t) * HEAD_DIM_C + d;
    vb[o] = (bf16_t)src[D_MODEL_C + QGROUPS * HEAD_DIM_C + e];
  }
}

// ---------------------------------------------------------------------------
// Flash attention (non-causal, full mask). Block = 128 q rows, 8 waves,
// each wave owns 16 q rows. Key tiles of 64. S and O via bf16 WMMA.
// ---------------------------------------------------------------------------
__global__ __launch_bounds__(256)
void flash_attn(const bf16_t* __restrict__ qb, const bf16_t* __restrict__ kb,
                const bf16_t* __restrict__ vb, bf16_t* __restrict__ ctx) {
  alignas(32) __shared__ bf16_t Qs[128][136];    // [q][d]
  alignas(32) __shared__ bf16_t Ks[64][136];     // [key][d]  (B-tile for QK^T)
  alignas(32) __shared__ bf16_t Vt[128][72];     // [d][key]  (B-tile for PV)
  alignas(32) __shared__ bf16_t Ps[8][16][72];   // per-wave P (A-tile for PV)

  const int tid   = threadIdx.x;
  const int wave  = tid >> 5;
  const int lane  = tid & 31;
  const int half8 = (lane >> 4) << 3;
  const int ln    = lane & 15;
  const int q0 = blockIdx.x * 128;
  const int h  = blockIdx.y;
  const int b  = blockIdx.z;
  const int g  = h >> 2;                 // HEADS_PER_GROUP = 4
  const size_t qbase  = ((size_t)(b * NUM_HEADS_C + h) * T_SZ + q0) * HEAD_DIM_C;
  const size_t kvbase = (size_t)(b * QGROUPS + g) * T_SZ * HEAD_DIM_C;

  // Load Q tile once: 128x128
  {
    const int r  = tid >> 4;
    const int d0 = (tid & 15) * 8;
#pragma unroll
    for (int p = 0; p < 8; ++p)
      *(bf16x8*)&Qs[r + p * 16][d0] =
          *(const bf16x8*)&qb[qbase + (size_t)(r + p * 16) * HEAD_DIM_C + d0];
  }

  f32x8 oacc[8] = {};
  float m_run[8], l_run[8];
#pragma unroll
  for (int r = 0; r < 8; ++r) { m_run[r] = -1e30f; l_run[r] = 0.f; }

  for (int k0 = 0; k0 < T_SZ; k0 += 64) {
    __syncthreads();  // previous iteration done reading Ks/Vt
    {
      const int r  = tid >> 4;
      const int d0 = (tid & 15) * 8;
#pragma unroll
      for (int p = 0; p < 4; ++p)
        *(bf16x8*)&Ks[r + p * 16][d0] =
            *(const bf16x8*)&kb[kvbase + (size_t)(k0 + r + p * 16) * HEAD_DIM_C + d0];
#pragma unroll
      for (int p = 0; p < 4; ++p) {
        union { bf16x8 v; bf16_t e[8]; } u;
        u.v = *(const bf16x8*)&vb[kvbase + (size_t)(k0 + r + p * 16) * HEAD_DIM_C + d0];
#pragma unroll
        for (int j = 0; j < 8; ++j) Vt[d0 + j][r + p * 16] = u.e[j];
      }
      if (k0 + 64 < T_SZ) {  // prefetch next K/V tile
        __builtin_prefetch(&kb[kvbase + (size_t)(k0 + 64 + r) * HEAD_DIM_C + d0], 0, 1);
        __builtin_prefetch(&vb[kvbase + (size_t)(k0 + 64 + r) * HEAD_DIM_C + d0], 0, 1);
      }
    }
    __syncthreads();

    // S = Q K^T : per wave 16(q) x 64(key), K-dim 128 in 4 chunks of 32
    bf16x16 afr[4];
#pragma unroll
    for (int kc = 0; kc < 4; ++kc)
      afr[kc] = load_frag_a(&Qs[wave * 16][kc * 32], 136);

    f32x8 sf[4];
#pragma unroll
    for (int nk = 0; nk < 4; ++nk) {
      f32x8 s = {};
#pragma unroll
      for (int kc = 0; kc < 4; ++kc) {
        bf16x16 bbf = load_frag_b(&Ks[nk * 16][kc * 32], 136);
        s = wmma_bf16(afr[kc], bbf, s);
      }
      sf[nk] = s * SCALE_C;
    }

    // Online softmax (per q row; rows live per-lane-half, per-VGPR-index)
#pragma unroll
    for (int r = 0; r < 8; ++r) {
      float v = fmaxf(fmaxf(sf[0][r], sf[1][r]), fmaxf(sf[2][r], sf[3][r]));
      v = fmaxf(v, __shfl_xor(v, 1, 32));
      v = fmaxf(v, __shfl_xor(v, 2, 32));
      v = fmaxf(v, __shfl_xor(v, 4, 32));
      v = fmaxf(v, __shfl_xor(v, 8, 32));
      const float mnew  = fmaxf(m_run[r], v);
      const float alpha = __expf(m_run[r] - mnew);
      m_run[r] = mnew;
      float ps = 0.f;
#pragma unroll
      for (int nk = 0; nk < 4; ++nk) {
        const float p = __expf(sf[nk][r] - mnew);
        ps += p;
        Ps[wave][r + half8][nk * 16 + ln] = (bf16_t)p;
      }
      ps += __shfl_xor(ps, 1, 32);
      ps += __shfl_xor(ps, 2, 32);
      ps += __shfl_xor(ps, 4, 32);
      ps += __shfl_xor(ps, 8, 32);
      l_run[r] = l_run[r] * alpha + ps;
#pragma unroll
      for (int f = 0; f < 8; ++f) oacc[f][r] *= alpha;
    }

    // O += P V : 8 dim-frags x 2 key-chunks (wave-local P in LDS, HW keeps
    // same-wave LDS ops in order)
#pragma unroll
    for (int f = 0; f < 8; ++f) {
#pragma unroll
      for (int kc = 0; kc < 2; ++kc) {
        bf16x16 a   = load_frag_a(&Ps[wave][0][kc * 32], 72);
        bf16x16 bbf = load_frag_b(&Vt[f * 16][kc * 32], 72);
        oacc[f] = wmma_bf16(a, bbf, oacc[f]);
      }
    }
  }

  // Normalize and write context in [B][T][H*128] layout
#pragma unroll
  for (int r = 0; r < 8; ++r) {
    const float inv_l = 1.0f / l_run[r];
    const int qrow = q0 + wave * 16 + r + half8;
    const size_t obase =
        ((size_t)(b * T_SZ) + qrow) * (size_t)D_MODEL_C + h * HEAD_DIM_C;
#pragma unroll
    for (int f = 0; f < 8; ++f)
      ctx[obase + f * 16 + ln] = (bf16_t)(oacc[f][r] * inv_l);
  }
}

// ---------------------------------------------------------------------------
// Host launcher
// ---------------------------------------------------------------------------
extern "C" void kernel_launch(void* const* d_in, const int* in_sizes, int n_in,
                              void* d_out, int out_size, void* d_ws, size_t ws_size,
                              hipStream_t stream) {
  (void)in_sizes; (void)n_in; (void)out_size; (void)ws_size;
  const float* x    = (const float*)d_in[0];
  const float* Wqkv = (const float*)d_in[1];
  const float* Wo   = (const float*)d_in[2];
  // d_in[3] = padding_mask, all-true in this problem -> ignored

  char* ws = (char*)d_ws;
  size_t off = 0;
  auto take = [&](size_t bytes) -> void* {
    void* p = ws + off;
    off += (bytes + 255) & ~(size_t)255;
    return p;
  };

  const size_t nx    = (size_t)B_SZ * T_SZ * D_MODEL_C;            // 16,777,216
  const size_t nwqkv = (size_t)D_MODEL_C * QKV_N;                  //  6,291,456
  const size_t nwo   = (size_t)D_MODEL_C * D_MODEL_C;              //  4,194,304
  const size_t nqkv  = (size_t)B_SZ * T_SZ * QKV_N;                // 25,165,824
  const size_t nkv   = (size_t)B_SZ * QGROUPS * T_SZ * HEAD_DIM_C; //  4,194,304

  bf16_t* xb    = (bf16_t*)take(nx * 2);
  bf16_t* wqkvb = (bf16_t*)take(nwqkv * 2);
  bf16_t* wob   = (bf16_t*)take(nwo * 2);
  float*  qkvf  = (float*) take(nqkv * 4);
  bf16_t* qb    = (bf16_t*)take(nx * 2);
  bf16_t* kbuf  = (bf16_t*)take(nkv * 2);
  bf16_t* vbuf  = (bf16_t*)take(nkv * 2);
  bf16_t* ctx   = (bf16_t*)take(nx * 2);

  cvt_f32_bf16<<<4096, 256, 0, stream>>>(x, xb, nx);
  cvt_f32_bf16<<<2048, 256, 0, stream>>>(Wqkv, wqkvb, nwqkv);
  cvt_f32_bf16<<<1024, 256, 0, stream>>>(Wo, wob, nwo);

  const int M = B_SZ * T_SZ;  // 8192
  gemm_bf16_f32<QKV_N, D_MODEL_C><<<dim3(M / 128, QKV_N / 128), 256, 0, stream>>>(
      xb, wqkvb, qkvf);

  rope_pack<<<B_SZ * T_SZ, 256, 0, stream>>>(qkvf, qb, kbuf, vbuf);

  flash_attn<<<dim3(T_SZ / 128, NUM_HEADS_C, B_SZ), 256, 0, stream>>>(
      qb, kbuf, vbuf, ctx);

  gemm_bf16_f32<D_MODEL_C, D_MODEL_C><<<dim3(M / 128, D_MODEL_C / 128), 256, 0, stream>>>(
      ctx, wob, (float*)d_out);
}
